// BitLinear_51230369907324
// MI455X (gfx1250) — matrix operand order repaired and checked by reading, not compiled, not run on hardware
//
#include <hip/hip_runtime.h>
#include <stdint.h>

typedef __attribute__((ext_vector_type(16))) _Float16 v16h;
typedef __attribute__((ext_vector_type(8)))  _Float16 v8h;
typedef __attribute__((ext_vector_type(8)))  float    v8f;
typedef __attribute__((ext_vector_type(4)))  float    v4f;

static constexpr int   K_DIM = 2048;
static constexpr int   N_DIM = 2048;
static constexpr int   M_DIM = 8192;   // 4 * 2048 flattened (batch, seq)
static constexpr float EPS_F = 1e-5f;

#define WS_SCALE_SLOT  1024                       // float index of scale in ws
#define WS_WQ_BYTE_OFF 8192                       // f16 ternary weights (8 MB)
#define WS_XH_BYTE_OFF (8192 + 8u * 1024 * 1024)  // f16 activations (32 MB)

// ---------------- Pass 1: partial sums of |W| ----------------
__global__ __launch_bounds__(256) void wabs_partial(const float* __restrict__ W,
                                                    float* __restrict__ partial) {
  __shared__ float sm[256];
  const int tid = threadIdx.x;
  const size_t total = (size_t)N_DIM * K_DIM;
  float s = 0.f;
  for (size_t i = (size_t)blockIdx.x * blockDim.x + tid; i < total;
       i += (size_t)gridDim.x * blockDim.x)
    s += fabsf(W[i]);
  sm[tid] = s;
  __syncthreads();
#pragma unroll
  for (int off = 128; off > 0; off >>= 1) {
    if (tid < off) sm[tid] += sm[tid + off];
    __syncthreads();
  }
  if (tid == 0) partial[blockIdx.x] = sm[0];
}

// ---------------- Pass 2: finalize scale ----------------
__global__ __launch_bounds__(1024) void wabs_final(float* __restrict__ ws) {
  __shared__ float sm[1024];
  const int tid = threadIdx.x;
  sm[tid] = ws[tid];
  __syncthreads();
#pragma unroll
  for (int off = 512; off > 0; off >>= 1) {
    if (tid < off) sm[tid] += sm[tid + off];
    __syncthreads();
  }
  if (tid == 0) {
    float mean = sm[0] * (1.0f / ((float)N_DIM * (float)K_DIM));
    ws[WS_SCALE_SLOT] = fmaxf(mean, EPS_F);
  }
}

// ---------------- Pass 3a: ternary-quantize W -> f16 (scale folded) ----------------
__global__ __launch_bounds__(256) void wquant(const float* __restrict__ W,
                                              const float* __restrict__ wsf,
                                              v8h* __restrict__ Wq) {
  const float scale  = wsf[WS_SCALE_SLOT];
  const float thresh = scale * (1.0f / 3.0f);
  const size_t i = (size_t)blockIdx.x * blockDim.x + threadIdx.x;  // 8 elems/thread
  const v4f* W4 = (const v4f*)W;
  v4f lo = W4[2 * i];
  v4f hi = W4[2 * i + 1];
  v8h q;
#pragma unroll
  for (int e = 0; e < 4; ++e) {
    float w0 = lo[e], w1 = hi[e];
    float q0 = (fabsf(w0) > thresh) ? (w0 > 0.f ? scale : -scale) : 0.f;
    float q1 = (fabsf(w1) > thresh) ? (w1 > 0.f ? scale : -scale) : 0.f;
    q[e]     = (_Float16)q0;
    q[e + 4] = (_Float16)q1;
  }
  Wq[i] = q;
}

// ---------------- Pass 3b: convert X f32 -> f16 ----------------
__global__ __launch_bounds__(256) void xcvt(const float* __restrict__ X,
                                            v8h* __restrict__ Xh) {
  const size_t i = (size_t)blockIdx.x * blockDim.x + threadIdx.x;  // 8 elems/thread
  const v4f* X4 = (const v4f*)X;
  v4f lo = X4[2 * i];
  v4f hi = X4[2 * i + 1];
  v8h h;
#pragma unroll
  for (int e = 0; e < 4; ++e) {
    h[e]     = (_Float16)lo[e];
    h[e + 4] = (_Float16)hi[e];
  }
  Xh[i] = h;
}

// ================= operand loaders =================
// A-operand (16x32 f16): lanes 0-15 hold K 0-15 of row (lane&15);
// lanes 16-31 hold K 16-31 contiguous.  Caller pre-offsets pointer by half*16.
__device__ __forceinline__ v16h load_a_f16(const _Float16* p) {
  v8h h0 = *(const v8h*)(p);
  v8h h1 = *(const v8h*)(p + 8);
  return __builtin_shufflevector(h0, h1, 0, 1, 2, 3, 4, 5, 6, 7,
                                 8, 9, 10, 11, 12, 13, 14, 15);
}

// B-operand (32x16 f16): lane = column (lane&15); K 0-15 (lanes<16) / 16-31 (lanes>=16)
// contiguous.  Caller pre-offsets pointer by half*16.
__device__ __forceinline__ v16h load_b_f16(const _Float16* p) {
  v8h h0 = *(const v8h*)(p);
  v8h h1 = *(const v8h*)(p + 8);
  return __builtin_shufflevector(h0, h1, 0, 1, 2, 3, 4, 5, 6, 7,
                                 8, 9, 10, 11, 12, 13, 14, 15);
}

// Fallback loaders (tiny-ws path): convert/quantize f32 sources in-register.
__device__ __forceinline__ v16h load_a_f32cvt(const float* p) {
  v4f f0 = *(const v4f*)(p + 0);
  v4f f1 = *(const v4f*)(p + 4);
  v4f f2 = *(const v4f*)(p + 8);
  v4f f3 = *(const v4f*)(p + 12);
  v16h r;
#pragma unroll
  for (int e = 0; e < 4; ++e) {
    r[e]      = (_Float16)f0[e];
    r[e + 4]  = (_Float16)f1[e];
    r[e + 8]  = (_Float16)f2[e];
    r[e + 12] = (_Float16)f3[e];
  }
  return r;
}

__device__ __forceinline__ v16h load_b_f32quant(const float* p, float scale,
                                                float thresh) {
  v16h r;
#pragma unroll
  for (int c = 0; c < 4; ++c) {
    v4f f = *(const v4f*)(p + 4 * c);
#pragma unroll
    for (int e = 0; e < 4; ++e) {
      float w = f[e];
      float q = (fabsf(w) > thresh) ? (w > 0.f ? scale : -scale) : 0.f;
      r[4 * c + e] = (_Float16)q;
    }
  }
  return r;
}

// ================= Pass 4: WMMA GEMM, pre-converted f16 operands =================
__global__ __launch_bounds__(256) void bitlinear_wmma_f16(
    const _Float16* __restrict__ Xh, const _Float16* __restrict__ Wq,
    float* __restrict__ Out) {
  const int lane = threadIdx.x & 31;   // wave32
  const int wave = threadIdx.x >> 5;   // 8 waves: 4 (M) x 2 (N)
  const int m0 = blockIdx.y * 128 + (wave >> 1) * 32;
  const int n0 = blockIdx.x * 64  + (wave & 1) * 32;
  const int r16  = lane & 15;
  const int half = lane >> 4;

  const _Float16* xr0 = Xh + (size_t)(m0      + r16) * K_DIM + half * 16;
  const _Float16* xr1 = Xh + (size_t)(m0 + 16 + r16) * K_DIM + half * 16;
  const _Float16* wr0 = Wq + (size_t)(n0      + r16) * K_DIM + half * 16;
  const _Float16* wr1 = Wq + (size_t)(n0 + 16 + r16) * K_DIM + half * 16;

  v8f c00 = {}, c01 = {}, c10 = {}, c11 = {};

#pragma unroll 2
  for (int k = 0; k < K_DIM; k += 32) {
    v16h a0 = load_a_f16(xr0 + k);
    v16h a1 = load_a_f16(xr1 + k);
    v16h b0 = load_b_f16(wr0 + k);
    v16h b1 = load_b_f16(wr1 + k);
    __builtin_prefetch(xr0 + k + 128, 0, 1);  // global_prefetch_b8
    __builtin_prefetch(wr0 + k + 128, 0, 1);
    c00 = __builtin_amdgcn_wmma_f32_16x16x32_f16(false, a0, false, b0,
                                                 (short)0, c00, false, false);
    c01 = __builtin_amdgcn_wmma_f32_16x16x32_f16(false, a0, false, b1,
                                                 (short)0, c01, false, false);
    c10 = __builtin_amdgcn_wmma_f32_16x16x32_f16(false, a1, false, b0,
                                                 (short)0, c10, false, false);
    c11 = __builtin_amdgcn_wmma_f32_16x16x32_f16(false, a1, false, b1,
                                                 (short)0, c11, false, false);
  }

  const int col   = lane & 15;
  const int rbase = 8 * half;
#pragma unroll
  for (int rr = 0; rr < 8; ++rr) {
    const size_t row0 = (size_t)(m0 + rbase + rr) * N_DIM;
    const size_t row1 = (size_t)(m0 + 16 + rbase + rr) * N_DIM;
    Out[row0 + n0 + col]      = c00[rr];
    Out[row0 + n0 + 16 + col] = c01[rr];
    Out[row1 + n0 + col]      = c10[rr];
    Out[row1 + n0 + 16 + col] = c11[rr];
  }
}

// ====== Pass 4 (fallback): fully fused GEMM from f32 sources (needs only scale) ======
__global__ __launch_bounds__(256) void bitlinear_wmma_fused(
    const float* __restrict__ X, const float* __restrict__ W,
    const float* __restrict__ wsf, float* __restrict__ Out) {
  const float scale  = wsf[WS_SCALE_SLOT];
  const float thresh = scale * (1.0f / 3.0f);

  const int lane = threadIdx.x & 31;
  const int wave = threadIdx.x >> 5;
  const int m0 = blockIdx.y * 128 + (wave >> 1) * 32;
  const int n0 = blockIdx.x * 64  + (wave & 1) * 32;
  const int r16  = lane & 15;
  const int half = lane >> 4;

  const float* xr0 = X + (size_t)(m0      + r16) * K_DIM + half * 16;
  const float* xr1 = X + (size_t)(m0 + 16 + r16) * K_DIM + half * 16;
  const float* wr0 = W + (size_t)(n0      + r16) * K_DIM + half * 16;
  const float* wr1 = W + (size_t)(n0 + 16 + r16) * K_DIM + half * 16;

  v8f c00 = {}, c01 = {}, c10 = {}, c11 = {};

#pragma unroll 2
  for (int k = 0; k < K_DIM; k += 32) {
    v16h a0 = load_a_f32cvt(xr0 + k);
    v16h a1 = load_a_f32cvt(xr1 + k);
    v16h b0 = load_b_f32quant(wr0 + k, scale, thresh);
    v16h b1 = load_b_f32quant(wr1 + k, scale, thresh);
    c00 = __builtin_amdgcn_wmma_f32_16x16x32_f16(false, a0, false, b0,
                                                 (short)0, c00, false, false);
    c01 = __builtin_amdgcn_wmma_f32_16x16x32_f16(false, a0, false, b1,
                                                 (short)0, c01, false, false);
    c10 = __builtin_amdgcn_wmma_f32_16x16x32_f16(false, a1, false, b0,
                                                 (short)0, c10, false, false);
    c11 = __builtin_amdgcn_wmma_f32_16x16x32_f16(false, a1, false, b1,
                                                 (short)0, c11, false, false);
  }

  const int col   = lane & 15;
  const int rbase = 8 * half;
#pragma unroll
  for (int rr = 0; rr < 8; ++rr) {
    const size_t row0 = (size_t)(m0 + rbase + rr) * N_DIM;
    const size_t row1 = (size_t)(m0 + 16 + rbase + rr) * N_DIM;
    Out[row0 + n0 + col]      = c00[rr];
    Out[row0 + n0 + 16 + col] = c01[rr];
    Out[row1 + n0 + col]      = c10[rr];
    Out[row1 + n0 + 16 + col] = c11[rr];
  }
}

extern "C" void kernel_launch(void* const* d_in, const int* in_sizes, int n_in,
                              void* d_out, int out_size, void* d_ws, size_t ws_size,
                              hipStream_t stream) {
  (void)in_sizes; (void)n_in; (void)out_size;
  const float* X = (const float*)d_in[0];   // (4, 2048, 2048) fp32
  const float* W = (const float*)d_in[1];   // (2048, 2048)    fp32
  float* Out = (float*)d_out;               // (4, 2048, 2048) fp32

  float* wsf = (float*)d_ws;  // [0..1023] partials, [1024] scale

  // Scale reduction always runs (needs ~4.1 KB of ws).
  wabs_partial<<<1024, 256, 0, stream>>>(W, wsf);
  wabs_final<<<1, 1024, 0, stream>>>(wsf);

  dim3 grid(N_DIM / 64, M_DIM / 128);  // 32 x 64 blocks, 256 threads each

  const size_t need_full =
      (size_t)WS_XH_BYTE_OFF + (size_t)M_DIM * K_DIM * sizeof(_Float16);

  if (ws_size >= need_full) {
    // Preferred path: pre-quantize W and pre-convert X to f16 once; GEMM inner
    // loop is pure b128 loads + v_wmma (all operands L2-resident: 8 + 32 MB).
    _Float16* Wq = (_Float16*)((char*)d_ws + WS_WQ_BYTE_OFF);
    _Float16* Xh = (_Float16*)((char*)d_ws + WS_XH_BYTE_OFF);
    wquant<<<(N_DIM * K_DIM / 8) / 256, 256, 0, stream>>>(W, wsf, (v8h*)Wq);
    xcvt<<<(M_DIM * K_DIM / 8) / 256, 256, 0, stream>>>(X, (v8h*)Xh);
    bitlinear_wmma_f16<<<grid, 256, 0, stream>>>(Xh, Wq, Out);
  } else {
    // Scratch-constrained fallback: quantize/convert in-register inside the GEMM.
    bitlinear_wmma_fused<<<grid, 256, 0, stream>>>(X, W, wsf, Out);
  }
}
